// LovaszSoftmax_83640193122776
// MI455X (gfx1250) — compile-verified, matrix operand order checked by reading, not run on hardware
//
#include <hip/hip_runtime.h>
#include <hip/hip_bf16.h>

// Lovasz-Softmax on MI455X (gfx1250, wave32).
// Sort replaced by a 2^17-bucket counting histogram per class (key = q16(err)*2 + gt),
// exploiting closed-form Lovasz-gradient telescoping within equal-key groups.
// All accumulation is integer-atomic or fixed-order fp64 => deterministic.

#define C_CLASSES 20
#define NKEY (1u << 17)                 // 65536 err buckets * 2 (gt bit in LSB)
#define CNT_SHIFT 43                    // bucket u64: [63:43]=count, [42:0]=sum of err*2^22
#define EMASK ((1ULL << 43) - 1ULL)
#define NB2 16                          // key ranges per class for the parallel scan
#define KEYS_PER_RANGE (NKEY / NB2)     // 8192
#define CHUNKS_PER_RANGE (KEYS_PER_RANGE / 256) // 32

typedef int v8i __attribute__((ext_vector_type(8)));

// ---------------- Kernel 1: softmax + packed histogram atomics ----------------
__global__ void softmax_hist_kernel(const float* __restrict__ logits,
                                    const int* __restrict__ targets,
                                    unsigned long long* __restrict__ hist,
                                    int N) {
  __shared__ float lds[256 * C_CLASSES];
  int r0 = blockIdx.x * 256;
  int rows = N - r0; if (rows > 256) rows = 256;
  int tot = rows * C_CLASSES;
  for (int i = threadIdx.x; i < tot; i += 256)
    lds[i] = logits[(long long)r0 * C_CLASSES + i];
  __syncthreads();
  int t = threadIdx.x;
  if (t < rows) {
    int row = r0 + t;
    float x[C_CLASSES];
    float m = -3.4e38f;
#pragma unroll
    for (int j = 0; j < C_CLASSES; ++j) { x[j] = lds[t * C_CLASSES + j]; m = fmaxf(m, x[j]); }
    float s = 0.f;
#pragma unroll
    for (int j = 0; j < C_CLASSES; ++j) { x[j] = __expf(x[j] - m); s += x[j]; }
    float inv = 1.f / s;
    int tgt = targets[row];
#pragma unroll
    for (int j = 0; j < C_CLASSES; ++j) {
      float p   = x[j] * inv;
      int   gt  = (j == tgt) ? 1 : 0;
      float err = fabsf((float)gt - p);          // in [0,1]
      unsigned q = (unsigned)(err * 65536.0f);
      if (q > 65535u) q = 65535u;
      unsigned key = (q << 1) | (unsigned)gt;
      unsigned long long efix = (unsigned long long)(err * 4194304.0f); // err*2^22, <= 2^22
      unsigned long long packed = (1ULL << CNT_SHIFT) | efix;           // no cross-field carry
      atomicAdd(&hist[(unsigned long long)j * NKEY + key], packed);
    }
  }
}

// ---------------- Kernel 2: per-class target counts via WMMA IU8 ----------------
// D = A(16x64 u8) x B(64x16 u8) + C.  B = all-ones bytes  =>  sum(D) = 16 * sum(A),
// independent of the VGPR striping of A/B. A carries one-hot indicator bytes for
// class c, so each v_wmma_i32_16x16x64_iu8 counts 1024 targets exactly in i32.
__global__ void target_count_wmma_kernel(const int* __restrict__ targets,
                                         unsigned* __restrict__ G, int N) {
  int c    = blockIdx.y;
  int lane = threadIdx.x & 31;
  int wave = blockIdx.x * (blockDim.x >> 5) + (threadIdx.x >> 5);
  int nWaves  = gridDim.x * (blockDim.x >> 5);
  int nChunks = (N + 1023) >> 10;
  v8i acc = {0, 0, 0, 0, 0, 0, 0, 0};
  const v8i ones = {0x01010101, 0x01010101, 0x01010101, 0x01010101,
                    0x01010101, 0x01010101, 0x01010101, 0x01010101};
  for (int ch = wave; ch < nChunks; ch += nWaves) {   // wave-uniform loop: EXEC all-ones at WMMA
    int base = ch << 10;
    v8i a;
    if (base + 1024 <= N) {
#pragma unroll
      for (int i = 0; i < 8; ++i) {
        const int4 tv = *(const int4*)(targets + base + i * 128 + lane * 4);
        int b = (tv.x == c ? 1 : 0) | ((tv.y == c ? 1 : 0) << 8) |
                ((tv.z == c ? 1 : 0) << 16) | ((tv.w == c ? 1 : 0) << 24);
        a[i] = b;
      }
    } else {
#pragma unroll
      for (int i = 0; i < 8; ++i) {
        int b = 0;
#pragma unroll
        for (int e = 0; e < 4; ++e) {
          int idx = base + i * 128 + lane * 4 + e;
          int v = (idx < N) ? targets[idx] : -1;
          b |= (v == c ? 1 : 0) << (8 * e);
        }
        a[i] = b;
      }
    }
    acc = __builtin_amdgcn_wmma_i32_16x16x64_iu8(false, a, false, ones, acc, false, false);
  }
  int s = 0;
#pragma unroll
  for (int i = 0; i < 8; ++i) s += acc[i];
#pragma unroll
  for (int m = 16; m; m >>= 1) s += __shfl_xor(s, m, 32);
  if (lane == 0 && s) atomicAdd(&G[c], (unsigned)(s >> 4));  // sum(D) = 16 * count
}

// ---------------- Kernel 3: per-range (cnt, gt) totals ----------------
__global__ void range_totals_kernel(const unsigned long long* __restrict__ hist,
                                    unsigned long long* __restrict__ blockTot) {
  int rb = blockIdx.x;   // rb = 0 covers the HIGHEST keys
  int c  = blockIdx.y;
  unsigned lo = NKEY - (unsigned)(rb + 1) * KEYS_PER_RANGE;
  unsigned cnt = 0, gt = 0;
  for (int i = threadIdx.x; i < KEYS_PER_RANGE; i += 256) {
    unsigned k = lo + (unsigned)i;
    unsigned long long v = hist[(unsigned long long)c * NKEY + k];
    unsigned m = (unsigned)(v >> CNT_SHIFT);
    cnt += m;
    gt  += (k & 1u) ? m : 0u;
  }
  __shared__ unsigned long long red[256];
  red[threadIdx.x] = ((unsigned long long)gt << 32) | cnt;
  __syncthreads();
  for (int s = 128; s > 0; s >>= 1) {
    if (threadIdx.x < s) red[threadIdx.x] += red[threadIdx.x + s];
    __syncthreads();
  }
  if (threadIdx.x == 0) blockTot[c * NB2 + rb] = red[0];
}

// ---------------- Kernel 4: tiny exclusive scan of range totals ----------------
__global__ void range_scan_kernel(const unsigned long long* __restrict__ blockTot,
                                  unsigned long long* __restrict__ base) {
  int c = threadIdx.x;
  if (c < C_CLASSES) {
    unsigned long long run = 0;
    for (int rb = 0; rb < NB2; ++rb) {
      base[c * NB2 + rb] = run;
      run += blockTot[c * NB2 + rb];
    }
  }
}

// ---------------- Kernel 5: descending bucket scan + Lovasz contributions ----------------
__global__ void lovasz_scan_kernel(const unsigned long long* __restrict__ hist,
                                   const unsigned long long* __restrict__ base,
                                   const unsigned* __restrict__ Gv,
                                   double* __restrict__ partial) {
  int rb = blockIdx.x, c = blockIdx.y;
  int tid = threadIdx.x, lane = tid & 31, wave = tid >> 5;
  unsigned long long b = base[c * NB2 + rb];
  unsigned R = (unsigned)(b & 0xffffffffu);   // elements above this range (descending order)
  unsigned S = (unsigned)(b >> 32);           // gt elements above this range
  unsigned Gu = Gv[c];
  double   Gd = (double)Gu;
  double   acc = 0.0;
  __shared__ unsigned wTotC[8], wTotG[8];
  unsigned topKey = NKEY - 1u - (unsigned)rb * KEYS_PER_RANGE;
  for (int ci = 0; ci < CHUNKS_PER_RANGE; ++ci) {
    unsigned k = topKey - (unsigned)(ci * 256 + tid);
    unsigned long long v = hist[(unsigned long long)c * NKEY + k];
    unsigned m   = (unsigned)(v >> CNT_SHIFT);
    unsigned gtm = (k & 1u) ? m : 0u;
    // wave32 inclusive scans of (m, gtm)
    unsigned ic = m, ig = gtm;
#pragma unroll
    for (int off = 1; off < 32; off <<= 1) {
      unsigned nc = __shfl_up(ic, (unsigned)off, 32);
      unsigned ng = __shfl_up(ig, (unsigned)off, 32);
      if (lane >= off) { ic += nc; ig += ng; }
    }
    if (lane == 31) { wTotC[wave] = ic; wTotG[wave] = ig; }
    __syncthreads();
    unsigned priorC = 0, priorG = 0, totC = 0, totG = 0;
#pragma unroll
    for (int w = 0; w < 8; ++w) {
      unsigned tc = wTotC[w], tg = wTotG[w];
      totC += tc; totG += tg;
      if (w < wave) { priorC += tc; priorG += tg; }
    }
    if (m) {
      unsigned r0 = R + priorC + (ic - m);        // descending rank of group start
      unsigned s0 = S + priorG + (ig - gtm);      // gt-count strictly above group
      double esum = (double)(v & EMASK) * (1.0 / 4194304.0);
      if (k & 1u) {
        // gt=1 group: union constant u = G + r0 - s0, every grad = 1/u (exact)
        double u = Gd + (double)r0 - (double)s0;
        acc += esum / u;
      } else {
        // gt=0 group: telescoped J(r0+m-1,s0) - J(r0-1,s0); J(-1,0)=0
        double num = Gd - (double)s0;
        double Jhi = 1.0 - num / (Gd + (double)r0 + (double)m - (double)s0);
        double Jlo = (r0 == 0) ? 0.0 : 1.0 - num / (Gd + (double)r0 - (double)s0);
        acc += (esum / (double)m) * (Jhi - Jlo);
      }
    }
    R += totC; S += totG;
    __syncthreads();
  }
  __shared__ double dred[256];
  dred[tid] = acc;
  __syncthreads();
  for (int s = 128; s > 0; s >>= 1) {
    if (tid < s) dred[tid] += dred[tid + s];
    __syncthreads();
  }
  if (tid == 0) partial[c * NB2 + rb] = dred[0];
}

// ---------------- Kernel 6: deterministic fixed-order finalize ----------------
__global__ void finalize_kernel(const double* __restrict__ partial, float* __restrict__ out) {
  if (threadIdx.x == 0 && blockIdx.x == 0) {
    double tot = 0.0;
    for (int c = 0; c < C_CLASSES; ++c) {
      double lc = 0.0;
      for (int rb = 0; rb < NB2; ++rb) lc += partial[c * NB2 + rb];
      tot += lc;
    }
    out[0] = (float)(tot / (double)C_CLASSES);
  }
}

extern "C" void kernel_launch(void* const* d_in, const int* in_sizes, int n_in,
                              void* d_out, int out_size, void* d_ws, size_t ws_size,
                              hipStream_t stream) {
  const float* logits  = (const float*)d_in[0];
  const int*   targets = (const int*)d_in[1];
  int N = in_sizes[1];

  size_t histBytes = (size_t)C_CLASSES * NKEY * 8ull;          // 20,971,520
  size_t offG    = histBytes;                                  // u32[32]
  size_t offTot  = offG + 128;                                 // u64[C*NB2]
  size_t offBase = offTot + (size_t)C_CLASSES * NB2 * 8ull;
  size_t offPart = offBase + (size_t)C_CLASSES * NB2 * 8ull;
  size_t needed  = offPart + (size_t)C_CLASSES * NB2 * 8ull;   // ~21 MB total
  if (ws_size < needed) return;

  char* ws = (char*)d_ws;
  unsigned long long* hist     = (unsigned long long*)ws;
  unsigned*           G        = (unsigned*)(ws + offG);
  unsigned long long* blockTot = (unsigned long long*)(ws + offTot);
  unsigned long long* base     = (unsigned long long*)(ws + offBase);
  double*             partial  = (double*)(ws + offPart);

  hipMemsetAsync(d_ws, 0, offG + 128, stream);  // zero hist + G every call (deterministic)

  int nb = (N + 255) / 256;
  softmax_hist_kernel<<<nb, 256, 0, stream>>>(logits, targets, hist, N);
  target_count_wmma_kernel<<<dim3(8, C_CLASSES), 256, 0, stream>>>(targets, G, N);
  range_totals_kernel<<<dim3(NB2, C_CLASSES), 256, 0, stream>>>(hist, blockTot);
  range_scan_kernel<<<1, 32, 0, stream>>>(blockTot, base);
  lovasz_scan_kernel<<<dim3(NB2, C_CLASSES), 256, 0, stream>>>(hist, base, G, partial);
  finalize_kernel<<<1, 32, 0, stream>>>(partial, (float*)d_out);
}
